// DiffusionConvGRU_30520037606076
// MI455X (gfx1250) — matrix operand order repaired and checked by reading, not compiled, not run on hardware
//
#include <hip/hip_runtime.h>
#include <hip/hip_bf16.h>

#define BG 8
#define NN 20000
#define EE 320000
#define FF 128
#define HH 128
#define LL 2
#define KK 3

typedef __attribute__((ext_vector_type(16))) _Float16 v16h;
typedef __attribute__((ext_vector_type(8)))  _Float16 v8h;
typedef __attribute__((ext_vector_type(8)))  float    v8f;

union V16 { v16h v; v8h h[2]; };

// ---------------- zero helper ----------------
__global__ void zero_f32(float* __restrict__ p, long long n) {
    long long i = (long long)blockIdx.x * blockDim.x + threadIdx.x;
    if (i < n) p[i] = 0.0f;
}

// ---------------- degree computation ----------------
__global__ void compute_deg(const int* __restrict__ src, const int* __restrict__ dst,
                            float* __restrict__ deg_out, float* __restrict__ deg_in) {
    long long i = (long long)blockIdx.x * blockDim.x + threadIdx.x;
    if (i >= (long long)BG * EE) return;
    int b = (int)(i / EE);
    unsafeAtomicAdd(&deg_out[(size_t)b * NN + src[i]], 1.0f);
    unsafeAtomicAdd(&deg_in [(size_t)b * NN + dst[i]], 1.0f);
}

// ---------------- hop prepare: xs = x * rsqrt(max(deg_out+k,1)); agg = k*xs ----------------
__global__ void hop_prepare(const float* __restrict__ x, const float* __restrict__ deg_out,
                            float khop, float* __restrict__ xs, float* __restrict__ agg) {
    long long idx = (long long)blockIdx.x * blockDim.x + threadIdx.x; // over B*N*32 float4s
    if (idx >= (long long)BG * NN * 32) return;
    long long node = idx >> 5;   // b*N + n
    int c4 = (int)(idx & 31);
    float rs = rsqrtf(fmaxf(deg_out[node] + khop, 1.0f));
    float4 v = ((const float4*)x)[node * 32 + c4];
    v.x *= rs; v.y *= rs; v.z *= rs; v.w *= rs;
    ((float4*)xs)[node * 32 + c4] = v;
    float4 a = make_float4(v.x * khop, v.y * khop, v.z * khop, v.w * khop);
    ((float4*)agg)[node * 32 + c4] = a;
}

// ---------------- hop scatter: agg[dst] += xs[src] (SpMM) ----------------
__global__ void hop_scatter(const int* __restrict__ src, const int* __restrict__ dst,
                            const float* __restrict__ xs, float* __restrict__ agg) {
    long long idx = (long long)blockIdx.x * blockDim.x + threadIdx.x; // E*32 per graph
    if (idx >= (long long)EE * 32) return;
    int b  = blockIdx.y;
    int e  = (int)(idx >> 5);
    int c4 = (int)(idx & 31);
    int s = src[(size_t)b * EE + e];
    int d = dst[(size_t)b * EE + e];
    float4 v = ((const float4*)xs)[((size_t)b * NN + s) * 32 + c4];
    float* ap = agg + ((size_t)b * NN + d) * FF + c4 * 4;
    unsafeAtomicAdd(ap + 0, v.x);
    unsafeAtomicAdd(ap + 1, v.y);
    unsafeAtomicAdd(ap + 2, v.z);
    unsafeAtomicAdd(ap + 3, v.w);
}

// ---------------- WMMA GEMM: C(b,row,col) = scale(A) @ B + bias ----------------
// A: (BG, NN, 128) f32. HAS_SCALE: per-row scale rsqrt(max(rowscale+khop,1)).
// TRANSB==0 -> B element (k,col) at Bm[k*NC + col] ; TRANSB==1 -> Bm[col*128 + k]
// NC is a template parameter so all row strides fold into instruction immediates.
// Workgroup: 256 threads = 8 waves; tile = 32 rows x 128 cols.
// Each wave: one 16-col slice, 2 row-tiles, B fragments cached in registers.
#define LDA 136   // half-element stride (272B) -> 16 lanes x b128 cover all 64 banks
template <int TRANSB, int HAS_SCALE, int NC>
__global__ void __launch_bounds__(256)
gemm128_wmma(const float* __restrict__ A, const float* __restrict__ Bm,
             const float* __restrict__ bias, float* __restrict__ C,
             const float* __restrict__ rowscale, float khop) {
    __shared__ _Float16 As[32 * LDA];    // As[m][k]
    __shared__ _Float16 Bs[128 * LDA];   // Bs[col][k] (col within this 128-col block)

    const int b = blockIdx.z;
    const int rowBase = blockIdx.x * 32;
    const int colBlock = blockIdx.y * 128;
    const float* Ab = A + ((size_t)b * NN + rowBase) * 128;
    float* Cb = C + ((size_t)b * NN + rowBase) * NC;

    // Stage scaled A tile (32x128) to LDS as f16
    for (int i = threadIdx.x; i < 32 * 128; i += 256) {
        int m = i >> 7, k = i & 127;
        float v = Ab[m * 128 + k];
        if (HAS_SCALE) {
            float d = rowscale[(size_t)b * NN + rowBase + m] + khop;
            v *= rsqrtf(fmaxf(d, 1.0f));
        }
        As[m * LDA + k] = (_Float16)v;
    }
    // Stage B panel (128 cols x 128 k) to LDS as f16, col-major rows
    if (TRANSB) {
        for (int i = threadIdx.x; i < 128 * 128; i += 256) {
            int cc = i >> 7, k = i & 127;                     // coalesced in k
            Bs[cc * LDA + k] = (_Float16)Bm[(colBlock + cc) * 128 + k];
        }
    } else {
        for (int i = threadIdx.x; i < 128 * 128; i += 256) {
            int k = i >> 7, cc = i & 127;                     // coalesced in cc
            Bs[cc * LDA + k] = (_Float16)Bm[k * NC + colBlock + cc];
        }
    }
    __syncthreads();

    const int lane = threadIdx.x & 31;
    const int wave = threadIdx.x >> 5;
    const int n  = lane & 15;   // col within 16-wide tile / row M for A frag
    const int hi = lane >> 4;   // half-wave select
    const int col = colBlock + wave * 16 + n;

    // B fragments for all 4 K-chunks, reused across both row tiles.
    // Layout (32x16 f16): lanes0-15 col=lane K=0..15; lanes16-31 col=lane-16 K=16..31
    V16 bf[4];
#pragma unroll
    for (int c = 0; c < 4; ++c) {
        const _Float16* p = &Bs[(wave * 16 + n) * LDA + c * 32 + hi * 16];
        bf[c].h[0] = *(const v8h*)(p);
        bf[c].h[1] = *(const v8h*)(p + 8);
    }

    const float bv = bias[col];
    // One base pointer per wave; all row offsets are compile-time (NC known)
    float* Cw = Cb + (size_t)hi * 8 * NC + col;
#pragma unroll
    for (int mt = 0; mt < 2; ++mt) {
        v8f acc;
#pragma unroll
        for (int r = 0; r < 8; ++r) acc[r] = bv;
#pragma unroll
        for (int c = 0; c < 4; ++c) {
            // A 16x32 f16 layout: lanes0-15 M=lane, K=0..15; lanes16-31 M=lane-16, K=16..31
            const _Float16* p = &As[(mt * 16 + n) * LDA + c * 32 + hi * 16];
            V16 a;
            a.h[0] = *(const v8h*)(p);
            a.h[1] = *(const v8h*)(p + 8);
            acc = __builtin_amdgcn_wmma_f32_16x16x32_f16(false, a.v, false, bf[c].v,
                                                         (short)0, acc, false, false);
        }
        // C/D layout: VGPR r -> M = r + 8*hi ; col = n
#pragma unroll
        for (int r = 0; r < 8; ++r)
            Cw[(mt * 16 + r) * NC] = acc[r];   // constant offsets -> store immediates
    }
}

// ---------------- sequential GRU scan (one workgroup per graph, 384 threads) ----------------
// gi: (BG, NN, 384) precomputed x@Wi^T + bi. Wh: (384,128). bhh: (384).
// WhT LDS layout [k/4][j][4] -> per-step GEMV = 32 x ds_load_b128 + float4 h broadcast.
__global__ void __launch_bounds__(384)
gru_seq(const float* __restrict__ gi, const float* __restrict__ Wh,
        const float* __restrict__ bhh, float* __restrict__ seq_out,
        float* __restrict__ out_final, int write_seq) {
    extern __shared__ float smem[];
    float* WhT = smem;                 // 128*384 floats, layout [k4][j][4]
    float* h   = smem + 128 * 384;     // [128] (16B aligned)
    float* gh  = h + 128;              // [384]

    int b = blockIdx.x;
    int j = threadIdx.x;               // 0..383

    for (int k = 0; k < 128; ++k)
        WhT[(((k >> 2) * 384) + j) * 4 + (k & 3)] = Wh[(size_t)j * 128 + k];
    float bj = bhh[j];
    if (j < 128) h[j] = 0.0f;
    __syncthreads();

    const float*  gib = gi + (size_t)b * NN * 384;
    float* seqb = seq_out ? seq_out + (size_t)b * NN * 128 : nullptr;
    const float4* Wv = (const float4*)WhT;   // index k4*384 + j
    const float4* h4 = (const float4*)h;

    for (int t = 0; t < NN; ++t) {
        // issue gi loads early; overlapped with the GEMV
        float giv  = (j < 256) ? gib[(size_t)t * 384 + j] : 0.0f;
        float ginn = (j < 128) ? gib[(size_t)t * 384 + 256 + j] : 0.0f;

        float acc = bj + giv;          // j>=256 rows keep only hidden part + bias
#pragma unroll 4
        for (int k4 = 0; k4 < 32; ++k4) {
            float4 w  = Wv[k4 * 384 + j];
            float4 hv = h4[k4];
            acc = fmaf(w.x, hv.x, acc);
            acc = fmaf(w.y, hv.y, acc);
            acc = fmaf(w.z, hv.z, acc);
            acc = fmaf(w.w, hv.w, acc);
        }
        gh[j] = acc;
        __syncthreads();

        if (j < 128) {
            float r  = 1.0f / (1.0f + __expf(-gh[j]));
            float z  = 1.0f / (1.0f + __expf(-gh[128 + j]));
            float nn = tanhf(ginn + r * gh[256 + j]);
            float h2 = (1.0f - z) * nn + z * h[j];
            h[j] = h2;                 // only thread j touches h[j] in this phase
            if (write_seq) seqb[(size_t)t * 128 + j] = h2;
        }
        __syncthreads();               // protects h (next GEMV) and gh (next overwrite)
    }
    if (!write_seq && j < 128)
        out_final[(size_t)b * 128 + j] = fmaxf(h[j], 0.0f);
}

// ---------------- host side ----------------
extern "C" void kernel_launch(void* const* d_in, const int* in_sizes, int n_in,
                              void* d_out, int out_size, void* d_ws, size_t ws_size,
                              hipStream_t stream) {
    const float* features = (const float*)d_in[0];   // (B,N,F)
    const int*   src      = (const int*)  d_in[1];   // (B,E)
    const int*   dst      = (const int*)  d_in[2];   // (B,E)
    const float* W        = (const float*)d_in[3];   // (F,H)
    const float* bvec     = (const float*)d_in[4];   // (H)
    const float* gWih     = (const float*)d_in[5];   // (L,3H,H)
    const float* gWhh     = (const float*)d_in[6];   // (L,3H,H)
    const float* gbih     = (const float*)d_in[7];   // (L,3H)
    const float* gbhh     = (const float*)d_in[8];   // (L,3H)
    float* out = (float*)d_out;

    float* ws = (float*)d_ws;
    const size_t BN  = (size_t)BG * NN;
    float* deg_out = ws;                       // BN
    float* deg_in  = deg_out + BN;             // BN
    float* xs      = deg_in  + BN;             // BN*128
    float* agg     = xs  + BN * 128;           // BN*128
    float* xbuf    = agg + BN * 128;           // BN*128
    float* gi      = xbuf + BN * 128;          // BN*384
    float* seq     = gi  + BN * 384;           // BN*128

    // degrees (re-zeroed every call for graph-replay determinism)
    {
        long long n = 2LL * BN;
        zero_f32<<<(unsigned)((n + 255) / 256), 256, 0, stream>>>(deg_out, n);
        long long ne = (long long)BG * EE;
        compute_deg<<<(unsigned)((ne + 255) / 256), 256, 0, stream>>>(src, dst, deg_out, deg_in);
    }

    // 3 graph-conv hops
    for (int k = 0; k < KK; ++k) {
        const float* xin = (k == 0) ? features : xbuf;
        long long np = (long long)BG * NN * 32;
        hop_prepare<<<(unsigned)((np + 255) / 256), 256, 0, stream>>>(
            xin, deg_out, (float)k, xs, agg);
        dim3 gs((unsigned)(((long long)EE * 32 + 255) / 256), BG);
        hop_scatter<<<gs, 256, 0, stream>>>(src, dst, xs, agg);
        dim3 gg(NN / 32, 1, BG);
        gemm128_wmma<0, 1, HH><<<gg, 256, 0, stream>>>(agg, W, bvec, xbuf,
                                                       deg_in, (float)k);
    }

    // GRU layers
    const size_t gru_smem = (size_t)(128 * 384 + 128 + 384) * sizeof(float); // ~194 KB (<320 KB/WGP)
    for (int l = 0; l < LL; ++l) {
        const float* Wi = gWih + (size_t)l * 3 * HH * HH;
        const float* Wh = gWhh + (size_t)l * 3 * HH * HH;
        const float* bi = gbih + (size_t)l * 3 * HH;
        const float* bh = gbhh + (size_t)l * 3 * HH;
        const float* xin = (l == 0) ? xbuf : seq;
        dim3 gg(NN / 32, 3, BG);   // 384 output cols = 3 col-blocks of 128
        gemm128_wmma<1, 0, 3 * HH><<<gg, 256, 0, stream>>>(xin, Wi, bi, gi,
                                                           nullptr, 0.0f);
        if (l < LL - 1)
            gru_seq<<<BG, 384, gru_smem, stream>>>(gi, Wh, bh, seq, nullptr, 1);
        else
            gru_seq<<<BG, 384, gru_smem, stream>>>(gi, Wh, bh, nullptr, out, 0);
    }
}
